// S5SSM_76785425317995
// MI455X (gfx1250) — matrix operand order repaired and checked by reading, not compiled, not run on hardware
//
#include <hip/hip_runtime.h>
#include <hip/hip_bf16.h>
#include <math.h>

// ---------------------------------------------------------------------------
// S5 SSM forward for MI455X (gfx1250, wave32, WMMA bf16 16x16x32, f32 accum)
//   L=16384 H=512 P=256 (complex), chunked parallel scan, GEMMs via v_wmma.
//   2x2 register-blocked macro-tiles (32x32 per wave) for WMMA density.
// ---------------------------------------------------------------------------

#define LSEQ   16384
#define HDIM   512
#define PDIM   256
#define CHUNK  128
#define NCHUNK (LSEQ / CHUNK)   // 128
#define KDIM2  (2 * PDIM)       // 512 : K of output GEMM ([Xr|Xi] concat)

typedef __attribute__((ext_vector_type(16))) __bf16 v16bf;
typedef __attribute__((ext_vector_type(8)))  __bf16 v8bf;
typedef __attribute__((ext_vector_type(4)))  __bf16 v4bf;
typedef __attribute__((ext_vector_type(8)))  float  v8f;

static __device__ __forceinline__ v16bf cat8(v8bf lo, v8bf hi) {
    return __builtin_shufflevector(lo, hi, 0,1,2,3,4,5,6,7,8,9,10,11,12,13,14,15);
}
static __device__ __forceinline__ v8bf ld8(const __bf16* p) {
    return *reinterpret_cast<const v8bf*>(p);
}

static __device__ __forceinline__ float2 cmul(float2 a, float2 b) {
    return make_float2(a.x * b.x - a.y * b.y, a.x * b.y + a.y * b.x);
}
// a*x + b (complex fma)
static __device__ __forceinline__ float2 cfma(float2 a, float2 x, float2 b) {
    return make_float2(a.x * x.x - a.y * x.y + b.x,
                       a.x * x.y + a.y * x.x + b.y);
}

// ---------------------------------------------------------------------------
// Prep 1: convert u (L,H) f32 -> bf16 row-major (A operand of input GEMM).
// ---------------------------------------------------------------------------
__global__ void s5_prep_u(const float* __restrict__ U, __bf16* __restrict__ Ubf) {
    const size_t i4 = (size_t)blockIdx.x * blockDim.x + threadIdx.x;
    const float4 v = reinterpret_cast<const float4*>(U)[i4];
    v4bf o;
    o[0] = (__bf16)v.x; o[1] = (__bf16)v.y; o[2] = (__bf16)v.z; o[3] = (__bf16)v.w;
    reinterpret_cast<v4bf*>(Ubf)[i4] = o;
}

// ---------------------------------------------------------------------------
// Prep 2: Bbf = [B_re ; B_im] (2*P,H) bf16;  Ccomb (H, 2P) bf16 with the
// conj-sym 2x / -2x folded in;  step = exp(log_step).
// ---------------------------------------------------------------------------
__global__ void s5_prep_params(const float* __restrict__ Bre, const float* __restrict__ Bim,
                               const float* __restrict__ Cre, const float* __restrict__ Cim,
                               const float* __restrict__ logstep,
                               __bf16* __restrict__ Bbf, __bf16* __restrict__ Ccomb,
                               float* __restrict__ step) {
    const int idx = blockIdx.x * blockDim.x + threadIdx.x;   // [0, 2*P*H)
    if (idx < PDIM * HDIM) {
        Bbf[idx]               = (__bf16)Bre[idx];
        Bbf[PDIM * HDIM + idx] = (__bf16)Bim[idx];
    }
    {   // Ccomb[h, q] : q<P -> 2*Cre[h,q] ; q>=P -> -2*Cim[h,q-P]
        const int h = idx / KDIM2, q = idx % KDIM2;
        const float v = (q < PDIM) ? 2.0f * Cre[h * PDIM + q]
                                   : -2.0f * Cim[h * PDIM + (q - PDIM)];
        Ccomb[idx] = (__bf16)v;
    }
    if (idx < PDIM) step[idx] = __expf(logstep[idx]);
}

// ---------------------------------------------------------------------------
// Input GEMM + ZOH discretization, fused.  2x2 blocking: each wave computes a
// 32(L) x 32(P) macro-tile for both re and im (8 WMMA / k-iter, 12 loads).
// ---------------------------------------------------------------------------
__global__ void s5_gemm_in(const __bf16* __restrict__ Ubf,
                           const __bf16* __restrict__ Bbf,   // [2][P][H]
                           const float*  __restrict__ ts,
                           const float*  __restrict__ lre,
                           const float*  __restrict__ lim,
                           const float*  __restrict__ step,
                           float2* __restrict__ Abar,        // (L,P)
                           float2* __restrict__ Bu) {        // (L,P)
    const int lane = threadIdx.x & 31;
    const int wid  = threadIdx.x >> 5;
    const int tile = blockIdx.x * 8 + wid;        // P-fast macro tiling
    const int tL   = (tile >> 3) << 5;            // P/32 == 8 macro tiles on p
    const int tP   = (tile & 7) << 5;
    const int n    = lane & 15;
    const int half = lane >> 4;

    const __bf16* ar0 = Ubf + (size_t)(tL + n) * HDIM + (half << 3);
    const __bf16* ar1 = ar0 + (size_t)16 * HDIM;
    const __bf16* bR0 = Bbf + (size_t)(tP + n) * HDIM + (half << 3);
    const __bf16* bR1 = bR0 + (size_t)16 * HDIM;
    const __bf16* bI0 = bR0 + (size_t)PDIM * HDIM;
    const __bf16* bI1 = bR1 + (size_t)PDIM * HDIM;

    v8f accR[2][2] = {{{}, {}}, {{}, {}}};   // [li][pj]
    v8f accI[2][2] = {{{}, {}}, {{}, {}}};
#pragma unroll
    for (int kb = 0; kb < HDIM / 32; ++kb) {
        const int k0 = kb * 32;
        const v16bf a0 = cat8(ld8(ar0 + k0), ld8(ar0 + k0 + 16));
        const v16bf a1 = cat8(ld8(ar1 + k0), ld8(ar1 + k0 + 16));
        const v16bf r0 = cat8(ld8(bR0 + k0), ld8(bR0 + k0 + 16));
        const v16bf r1 = cat8(ld8(bR1 + k0), ld8(bR1 + k0 + 16));
        const v16bf i0 = cat8(ld8(bI0 + k0), ld8(bI0 + k0 + 16));
        const v16bf i1 = cat8(ld8(bI1 + k0), ld8(bI1 + k0 + 16));
        accR[0][0] = __builtin_amdgcn_wmma_f32_16x16x32_bf16(false, a0, false, r0, (short)0, accR[0][0], false, false);
        accR[0][1] = __builtin_amdgcn_wmma_f32_16x16x32_bf16(false, a0, false, r1, (short)0, accR[0][1], false, false);
        accR[1][0] = __builtin_amdgcn_wmma_f32_16x16x32_bf16(false, a1, false, r0, (short)0, accR[1][0], false, false);
        accR[1][1] = __builtin_amdgcn_wmma_f32_16x16x32_bf16(false, a1, false, r1, (short)0, accR[1][1], false, false);
        accI[0][0] = __builtin_amdgcn_wmma_f32_16x16x32_bf16(false, a0, false, i0, (short)0, accI[0][0], false, false);
        accI[0][1] = __builtin_amdgcn_wmma_f32_16x16x32_bf16(false, a0, false, i1, (short)0, accI[0][1], false, false);
        accI[1][0] = __builtin_amdgcn_wmma_f32_16x16x32_bf16(false, a1, false, i0, (short)0, accI[1][0], false, false);
        accI[1][1] = __builtin_amdgcn_wmma_f32_16x16x32_bf16(false, a1, false, i1, (short)0, accI[1][1], false, false);
    }

    // --- fused ZOH discretization epilogue ---
    float sp[2], ar_[2], ai_[2], inv_[2];
#pragma unroll
    for (int pj = 0; pj < 2; ++pj) {
        const int p = tP + (pj << 4) + n;
        sp[pj] = step[p];
        ar_[pj] = lre[p];
        ai_[pj] = lim[p];
        inv_[pj] = 1.0f / (ar_[pj] * ar_[pj] + ai_[pj] * ai_[pj]);
    }
#pragma unroll
    for (int li = 0; li < 2; ++li) {
#pragma unroll
        for (int i = 0; i < 8; ++i) {
            const int   l   = tL + (li << 4) + i + (half << 3);   // C-frag row
            const float tsl = ts[l];
#pragma unroll
            for (int pj = 0; pj < 2; ++pj) {
                const int   p  = tP + (pj << 4) + n;
                const float dt = tsl * sp[pj];
                const float e  = __expf(ar_[pj] * dt);
                float s, c;
                __sincosf(ai_[pj] * dt, &s, &c);
                const float Are = e * c, Aim = e * s;
                // gamma = (Lambda_bar - 1) * conj(Lambda) / |Lambda|^2
                const float gr = ((Are - 1.0f) * ar_[pj] + Aim * ai_[pj]) * inv_[pj];
                const float gi = (Aim * ar_[pj] - (Are - 1.0f) * ai_[pj]) * inv_[pj];
                const float br = accR[li][pj][i], bi = accI[li][pj][i];
                const size_t o = (size_t)l * PDIM + p;
                Abar[o] = make_float2(Are, Aim);
                Bu[o]   = make_float2(gr * br - gi * bi, gr * bi + gi * br);
            }
        }
    }
}

// ---------------------------------------------------------------------------
// Scan phase 1: per-(chunk, mode) aggregate (prod A, folded b).
// ---------------------------------------------------------------------------
__global__ void s5_chunk_agg(const float2* __restrict__ A, const float2* __restrict__ B,
                             float2* __restrict__ Agg, float2* __restrict__ Bagg) {
    const int p = threadIdx.x, c = blockIdx.x;
    float2 Ap = make_float2(1.0f, 0.0f), bp = make_float2(0.0f, 0.0f);
    const int base = c * CHUNK;
    for (int t = 0; t < CHUNK; ++t) {
        const size_t o = (size_t)(base + t) * PDIM + p;
        const float2 At = A[o];
        bp = cfma(At, bp, B[o]);
        Ap = cmul(At, Ap);
    }
    Agg[c * PDIM + p]  = Ap;
    Bagg[c * PDIM + p] = bp;
}

// ---------------------------------------------------------------------------
// Scan phase 2: exclusive scan of chunk aggregates (serial over 128 chunks).
// ---------------------------------------------------------------------------
__global__ void s5_chunk_scan(const float2* __restrict__ Agg, const float2* __restrict__ Bagg,
                              float2* __restrict__ Pref) {
    const int p = threadIdx.x;
    float2 x = make_float2(0.0f, 0.0f);
    for (int c = 0; c < NCHUNK; ++c) {
        Pref[c * PDIM + p] = x;
        x = cfma(Agg[c * PDIM + p], x, Bagg[c * PDIM + p]);
    }
}

// ---------------------------------------------------------------------------
// Scan phase 3: re-scan each chunk from its prefix, emit Xcat = [Xr|Xi] bf16.
// ---------------------------------------------------------------------------
__global__ void s5_apply(const float2* __restrict__ A, const float2* __restrict__ B,
                         const float2* __restrict__ Pref, __bf16* __restrict__ Xcat) {
    const int p = threadIdx.x, c = blockIdx.x;
    float2 x = Pref[c * PDIM + p];
    const int base = c * CHUNK;
    for (int t = 0; t < CHUNK; ++t) {
        const int l = base + t;
        const size_t o = (size_t)l * PDIM + p;
        x = cfma(A[o], x, B[o]);
        Xcat[(size_t)l * KDIM2 + p]        = (__bf16)x.x;
        Xcat[(size_t)l * KDIM2 + PDIM + p] = (__bf16)x.y;
    }
}

// ---------------------------------------------------------------------------
// Output GEMM: y = Xcat @ Ccomb^T + D*u.  2x2 blocking: 32(L) x 32(H) per
// wave (4 WMMA / k-iter, 8 loads).
// ---------------------------------------------------------------------------
__global__ void s5_gemm_out(const __bf16* __restrict__ Xcat,
                            const __bf16* __restrict__ Ccomb,  // (H, 2P)
                            const float*  __restrict__ D,
                            const float*  __restrict__ U,
                            float* __restrict__ Y) {
    const int lane = threadIdx.x & 31;
    const int wid  = threadIdx.x >> 5;
    const int tile = blockIdx.x * 8 + wid;
    const int tL   = (tile >> 4) << 5;            // H/32 == 16 macro tiles on h
    const int tH   = (tile & 15) << 5;
    const int n    = lane & 15;
    const int half = lane >> 4;

    const __bf16* ar0 = Xcat  + (size_t)(tL + n) * KDIM2 + (half << 3);
    const __bf16* ar1 = ar0 + (size_t)16 * KDIM2;
    const __bf16* br0 = Ccomb + (size_t)(tH + n) * KDIM2 + (half << 3);
    const __bf16* br1 = br0 + (size_t)16 * KDIM2;

    v8f acc[2][2] = {{{}, {}}, {{}, {}}};        // [li][hj]
#pragma unroll
    for (int kb = 0; kb < KDIM2 / 32; ++kb) {
        const int k0 = kb * 32;
        const v16bf a0 = cat8(ld8(ar0 + k0), ld8(ar0 + k0 + 16));
        const v16bf a1 = cat8(ld8(ar1 + k0), ld8(ar1 + k0 + 16));
        const v16bf b0 = cat8(ld8(br0 + k0), ld8(br0 + k0 + 16));
        const v16bf b1 = cat8(ld8(br1 + k0), ld8(br1 + k0 + 16));
        acc[0][0] = __builtin_amdgcn_wmma_f32_16x16x32_bf16(false, a0, false, b0, (short)0, acc[0][0], false, false);
        acc[0][1] = __builtin_amdgcn_wmma_f32_16x16x32_bf16(false, a0, false, b1, (short)0, acc[0][1], false, false);
        acc[1][0] = __builtin_amdgcn_wmma_f32_16x16x32_bf16(false, a1, false, b0, (short)0, acc[1][0], false, false);
        acc[1][1] = __builtin_amdgcn_wmma_f32_16x16x32_bf16(false, a1, false, b1, (short)0, acc[1][1], false, false);
    }

#pragma unroll
    for (int hj = 0; hj < 2; ++hj) {
        const int   h = tH + (hj << 4) + n;
        const float d = D[h];
#pragma unroll
        for (int li = 0; li < 2; ++li) {
#pragma unroll
            for (int i = 0; i < 8; ++i) {
                const int l = tL + (li << 4) + i + (half << 3);
                const size_t o = (size_t)l * HDIM + h;
                Y[o] = acc[li][hj][i] + d * U[o];
            }
        }
    }
}

// ---------------------------------------------------------------------------
// Host launcher — all scratch carved from d_ws, everything on `stream`.
// ---------------------------------------------------------------------------
extern "C" void kernel_launch(void* const* d_in, const int* in_sizes, int n_in,
                              void* d_out, int out_size, void* d_ws, size_t ws_size,
                              hipStream_t stream) {
    (void)in_sizes; (void)n_in; (void)out_size; (void)ws_size;

    const float* U   = (const float*)d_in[0];   // (L,H)
    const float* TS  = (const float*)d_in[1];   // (L,)
    const float* LRE = (const float*)d_in[2];   // (P,)
    const float* LIM = (const float*)d_in[3];   // (P,)
    const float* BRE = (const float*)d_in[4];   // (P,H)
    const float* BIM = (const float*)d_in[5];   // (P,H)
    const float* CRE = (const float*)d_in[6];   // (H,P)
    const float* CIM = (const float*)d_in[7];   // (H,P)
    const float* DD  = (const float*)d_in[8];   // (H,)
    const float* LST = (const float*)d_in[9];   // (P,)
    float* Y = (float*)d_out;

    char* ws = (char*)d_ws;
    size_t off = 0;
    auto carve = [&](size_t bytes) -> char* {
        char* p = ws + off;
        off += (bytes + 255) & ~(size_t)255;
        return p;
    };
    __bf16* Ubf   = (__bf16*)carve((size_t)LSEQ * HDIM * 2);        // 16 MB
    __bf16* Bbf   = (__bf16*)carve((size_t)2 * PDIM * HDIM * 2);    // 0.5 MB
    __bf16* Ccomb = (__bf16*)carve((size_t)HDIM * KDIM2 * 2);       // 0.5 MB
    float2* Abar  = (float2*)carve((size_t)LSEQ * PDIM * 8);        // 32 MB
    float2* Bu    = (float2*)carve((size_t)LSEQ * PDIM * 8);        // 32 MB
    float2* Agg   = (float2*)carve((size_t)NCHUNK * PDIM * 8);
    float2* Bagg  = (float2*)carve((size_t)NCHUNK * PDIM * 8);
    float2* Pref  = (float2*)carve((size_t)NCHUNK * PDIM * 8);
    __bf16* Xcat  = (__bf16*)carve((size_t)LSEQ * KDIM2 * 2);       // 16 MB
    float*  Step  = (float*)carve((size_t)PDIM * 4);

    // Prep: bf16 operand staging
    s5_prep_u<<<(LSEQ * HDIM / 4) / 256, 256, 0, stream>>>(U, Ubf);
    s5_prep_params<<<(2 * PDIM * HDIM) / 256, 256, 0, stream>>>(
        BRE, BIM, CRE, CIM, LST, Bbf, Ccomb, Step);

    // Input GEMM + discretization (fused), 32x32 macro-tiles
    s5_gemm_in<<<(LSEQ / 32) * (PDIM / 32) / 8, 256, 0, stream>>>(
        Ubf, Bbf, TS, LRE, LIM, Step, Abar, Bu);

    // Chunked parallel scan
    s5_chunk_agg<<<NCHUNK, PDIM, 0, stream>>>(Abar, Bu, Agg, Bagg);
    s5_chunk_scan<<<1, PDIM, 0, stream>>>(Agg, Bagg, Pref);
    s5_apply<<<NCHUNK, PDIM, 0, stream>>>(Abar, Bu, Pref, Xcat);

    // Output GEMM + skip connection, 32x32 macro-tiles
    s5_gemm_out<<<(LSEQ / 32) * (HDIM / 32) / 8, 256, 0, stream>>>(
        Xcat, Ccomb, DD, U, Y);
}
